// LSTMSeqDecoder_52613349376571
// MI455X (gfx1250) — matrix operand order repaired and checked by reading, not compiled, not run on hardware
//
#include <hip/hip_runtime.h>
#include <hip/hip_bf16.h>

// Problem dims (fixed by the reference)
#define B_   64
#define TD_  128
#define TE_  128
#define D_   1024
#define ENC_ 1024
#define E_   32
#define L_   512
#define KZ   (E_ + D_ + D_)   // 2080 : concat row [x(32) | att(1024) | h(1024)]
#define KA   (D_ + ENC_)      // 2048 : concat row [h(1024) | ctx(1024)]
#define NZ   (4 * D_)         // 4096 gate width

using bf16 = __hip_bfloat16;

typedef __attribute__((ext_vector_type(16))) __bf16 v16bf;
typedef __attribute__((ext_vector_type(8)))  __bf16 v8bf;
typedef __attribute__((ext_vector_type(8)))  float  v8f;
typedef __attribute__((ext_vector_type(4)))  int    v4i_;

// CDNA5 async global->LDS path (ASYNCcnt-tracked), with safe fallback.
#if defined(__has_builtin)
#  if __has_builtin(__builtin_amdgcn_global_load_async_to_lds_b128) && \
      __has_builtin(__builtin_amdgcn_s_wait_asynccnt)
#    define HAVE_ASYNC_LDS 1
#  endif
#endif
#ifndef HAVE_ASYNC_LDS
#  define HAVE_ASYNC_LDS 0
#endif

#if HAVE_ASYNC_LDS
typedef __attribute__((address_space(1))) v4i_ g_v4i;   // global int4
typedef __attribute__((address_space(3))) v4i_ l_v4i;   // LDS int4
#endif

__device__ __forceinline__ float sigm(float x) { return 1.f / (1.f + __expf(-x)); }

// ---------------------------------------------------------------------------
// One-time prep: transpose fp32 weights into bf16 [N][K] so each lane's WMMA
// B fragment (16 contiguous K values at fixed N) is 2x b128 loads.
// ---------------------------------------------------------------------------
__global__ void k_tr_bf16(const float* __restrict__ W, bf16* __restrict__ WT,
                          int K, int N, long long total) {
    long long i = (long long)blockIdx.x * 256 + threadIdx.x;
    if (i >= total) return;
    int n = (int)(i / K), k = (int)(i % K);
    WT[i] = __float2bfloat16(W[(long long)k * N + n]);
}

// Fused [Wk;Wr] -> WzT[4096][2080]
__global__ void k_wz_tr(const float* __restrict__ Wk, const float* __restrict__ Wr,
                        bf16* __restrict__ WzT) {
    long long i = (long long)blockIdx.x * 256 + threadIdx.x;
    if (i >= (long long)NZ * KZ) return;
    int n = (int)(i / KZ), k = (int)(i % KZ);
    float v = (k < E_ + D_) ? Wk[(long long)k * NZ + n]
                            : Wr[(long long)(k - (E_ + D_)) * NZ + n];
    WzT[i] = __float2bfloat16(v);
}

__global__ void k_cvt_bf16(const float* __restrict__ X, bf16* __restrict__ Y,
                           long long total) {
    long long i = (long long)blockIdx.x * 256 + threadIdx.x;
    if (i >= total) return;
    Y[i] = __float2bfloat16(X[i]);
}

// All teacher-forced inputs at once: xall[t][b][j] = emb[tokens[b][t]][j]
__global__ void k_embed_all(const int* __restrict__ tokens, const float* __restrict__ emb,
                            bf16* __restrict__ xall) {
    int i = blockIdx.x * 256 + threadIdx.x;  // B_*TD_*E_
    if (i >= B_ * TD_ * E_) return;
    int j = i % E_;
    int bt = i / E_;
    int t = bt % TD_, b = bt / TD_;
    int tok = tokens[b * TD_ + t];
    xall[((long long)t * B_ + b) * E_ + j] = __float2bfloat16(emb[tok * E_ + j]);
}

__global__ void k_init_state(const float* __restrict__ h0, const float* __restrict__ c0,
                             float* __restrict__ h, float* __restrict__ c,
                             bf16* __restrict__ acat1) {
    int i = blockIdx.x * 256 + threadIdx.x;
    if (i >= B_ * D_) return;
    int b = i / D_, d = i % D_;
    h[i] = h0[i];
    c[i] = c0[i];
    acat1[(long long)b * KZ + E_ + d]      = __float2bfloat16(0.f);    // att0 = 0
    acat1[(long long)b * KZ + E_ + D_ + d] = __float2bfloat16(h0[i]);  // h0
}

// ---------------------------------------------------------------------------
// Wave32 WMMA GEMM: C[b] = A[b] (MTILES*16 x K bf16 row-major, lda) x BT^T
// (BT is [N][K] bf16). Block = 256 threads = 8 waves; wave owns 16 columns,
// full M. A tile (ROWS x 32) double-buffered in LDS (padded stride 40 ->
// conflict-free ds_load_b128); next tile prefetched (async-to-LDS if
// available) while current one is consumed. Optional AX: replaces the first
// 32 K-elements of each row (kk==0 chunk) -- fuses the embedding slice of
// the concat activation without a separate copy kernel.
// Epilogue: +bias[n], optional fp32 C and/or bf16 Cbf writeback.
// ---------------------------------------------------------------------------
template <int MTILES>
__global__ __launch_bounds__(256)
void k_gemm_bf16(const bf16* __restrict__ A_, long long aBatch, int lda,
                 const bf16* __restrict__ AX_,          // optional [ROWS][32] bf16
                 const bf16* __restrict__ BT_,
                 float* __restrict__ C, long long cBatch, int ldc,
                 bf16* __restrict__ Cbf_, int ldcbf,
                 const float* __restrict__ bias,
                 int K, int blocksPerBatch) {
    constexpr int ROWS   = MTILES * 16;
    constexpr int LDAS   = 40;                 // bf16 units; 80B row stride
    constexpr int BUFSZ  = ROWS * LDAS;
    constexpr int CHUNKS = (ROWS * 4) / 256;   // 16B chunks per thread (1 or 2)
    __shared__ __align__(16) __bf16 As[2 * BUFSZ];

    const int tid  = threadIdx.x;
    const int wave = tid >> 5;
    const int lane = tid & 31;
    const int ln   = lane & 15;   // N within tile / M row within tile
    const int hi   = lane >> 4;   // K-half selector
    const int batch = blockIdx.x / blocksPerBatch;
    const int nb    = blockIdx.x % blocksPerBatch;
    const int n0    = (nb * 8 + wave) * 16;

    const __bf16* Ab = reinterpret_cast<const __bf16*>(A_) + (long long)batch * aBatch;
    const __bf16* AX = reinterpret_cast<const __bf16*>(AX_);
    const __bf16* BT = reinterpret_cast<const __bf16*>(BT_);
    const __bf16* Bw = BT + (long long)(n0 + ln) * K + hi * 16;

    // Stage one ROWS x 32 A tile into LDS buffer `buf` for k-offset `kk`.
    auto stage = [&](int kk, int buf) {
        __bf16* dst = &As[buf * BUFSZ];
#pragma unroll
        for (int cchunk = 0; cchunk < CHUNKS; ++cchunk) {
            int i = tid + cchunk * 256;
            int r = i >> 2, s = i & 3;
            const __bf16* sp = (AX && kk == 0)
                                   ? (AX + r * E_ + s * 8)
                                   : (Ab + (long long)r * lda + kk + s * 8);
#if HAVE_ASYNC_LDS
            __builtin_amdgcn_global_load_async_to_lds_b128(
                (g_v4i*)sp, (l_v4i*)(dst + r * LDAS + s * 8), 0, 0);
#else
            *reinterpret_cast<uint4*>(dst + r * LDAS + s * 8) =
                *reinterpret_cast<const uint4*>(sp);
#endif
        }
    };

    v8f acc[MTILES] = {};

    stage(0, 0);
#if HAVE_ASYNC_LDS
    __builtin_amdgcn_s_wait_asynccnt(0);
#endif
    __syncthreads();

    const int iters = K / 32;
    for (int it = 0; it < iters; ++it) {
        const int kk  = it * 32;
        const int cur = it & 1;
        if (kk + 32 < K) stage(kk + 32, cur ^ 1);  // prefetch next tile

        // B fragment: column n0+ln, K = kk + hi*16 .. +15 (contiguous in BT)
        v16bf bfrag = *reinterpret_cast<const v16bf*>(Bw + kk);
        if (kk + 32 < K) __builtin_prefetch(Bw + kk + 32, 0, 0);

        const __bf16* Ac = &As[cur * BUFSZ];
#pragma unroll
        for (int m = 0; m < MTILES; ++m) {
            // A layout: lane<16 -> K 0-15 ; lane>=16 -> K 16-31
            const __bf16* ap = Ac + (m * 16 + ln) * LDAS + hi * 16;
            v8bf lo  = *reinterpret_cast<const v8bf*>(ap);
            v8bf hi8 = *reinterpret_cast<const v8bf*>(ap + 8);
            v16bf af = __builtin_shufflevector(lo, hi8, 0, 1, 2, 3, 4, 5, 6, 7,
                                               8, 9, 10, 11, 12, 13, 14, 15);
            acc[m] = __builtin_amdgcn_wmma_f32_16x16x32_bf16(
                false, af, false, bfrag, (short)0, acc[m], false, false);
        }

#if HAVE_ASYNC_LDS
        __builtin_amdgcn_s_wait_asynccnt(0);  // next tile landed in LDS
#endif
        __syncthreads();                      // everyone done with `cur`, next visible
    }

    const int col = n0 + ln;
    const float bb = bias ? bias[col] : 0.f;
    bf16* Cbf = Cbf_;
#pragma unroll
    for (int m = 0; m < MTILES; ++m) {
#pragma unroll
        for (int r = 0; r < 8; ++r) {
            int row = m * 16 + hi * 8 + r;  // D layout: VGPR r -> M = hi*8 + r
            float v = acc[m][r] + bb;
            if (C)   C[(long long)batch * cBatch + (long long)row * ldc + col] = v;
            if (Cbf) Cbf[(long long)row * ldcbf + col] = __float2bfloat16(v);
        }
    }
}

// ---------------------------------------------------------------------------
// LSTM cell elementwise: gate order i,f,g,o (keras). Writes fp32 h state and
// bf16 h into acat1 h-slice for the next step's gate GEMM.
// ---------------------------------------------------------------------------
__global__ void k_cell(const float* __restrict__ z, float* __restrict__ h,
                       float* __restrict__ c, bf16* __restrict__ acat1) {
    int idx = blockIdx.x * 256 + threadIdx.x;
    if (idx >= B_ * D_) return;
    int b = idx / D_, d = idx % D_;
    const float* zb = z + (long long)b * NZ;
    float ig = sigm(zb[d]);
    float fg = sigm(zb[d + D_]);
    float gg = tanhf(zb[d + 2 * D_]);
    float og = sigm(zb[d + 3 * D_]);
    float cn = fg * c[idx] + ig * gg;
    float hn = og * tanhf(cn);
    c[idx] = cn;
    h[idx] = hn;
    acat1[(long long)b * KZ + E_ + D_ + d] = __float2bfloat16(hn);
}

// ---------------------------------------------------------------------------
// Fused Luong attention: score = h . keys[b,t,:], softmax over TE,
// ctx = sum_t align * memory[b,t,:]; emits acat2 = bf16([h | ctx]).
// ~34 MFLOP/step -> VALU; block per batch row.
// ---------------------------------------------------------------------------
__global__ __launch_bounds__(128)
void k_attention(const float* __restrict__ h, const float* __restrict__ keys,
                 const float* __restrict__ memory, bf16* __restrict__ acat2) {
    __shared__ float hl[D_];
    __shared__ float sc[TE_];
    __shared__ float red[TE_];
    const int b = blockIdx.x, t = threadIdx.x;

    for (int d = t; d < D_; d += 128) hl[d] = h[b * D_ + d];
    __syncthreads();

    const float* kr = keys + ((long long)b * TE_ + t) * D_;
    float acc = 0.f;
    for (int d = 0; d < D_; d += 4) {
        float4 k4 = *reinterpret_cast<const float4*>(kr + d);
        acc += hl[d] * k4.x + hl[d + 1] * k4.y + hl[d + 2] * k4.z + hl[d + 3] * k4.w;
    }
    sc[t] = acc; red[t] = acc;
    __syncthreads();
    for (int s = 64; s > 0; s >>= 1) { if (t < s) red[t] = fmaxf(red[t], red[t + s]); __syncthreads(); }
    float mx = red[0];
    __syncthreads();
    float ex = __expf(sc[t] - mx);
    red[t] = ex; sc[t] = ex;
    __syncthreads();
    for (int s = 64; s > 0; s >>= 1) { if (t < s) red[t] += red[t + s]; __syncthreads(); }
    float inv = 1.f / red[0];
    __syncthreads();
    sc[t] *= inv;  // align[b][t]
    __syncthreads();

    for (int d = t; d < D_; d += 128)
        acat2[(long long)b * KA + d] = __float2bfloat16(hl[d]);

    const float* mb = memory + (long long)b * TE_ * ENC_;
    for (int e0 = t; e0 < ENC_; e0 += 128) {
        float cacc = 0.f;
        for (int tt = 0; tt < TE_; ++tt)
            cacc += sc[tt] * mb[(long long)tt * ENC_ + e0];
        acat2[(long long)b * KA + D_ + e0] = __float2bfloat16(cacc);
    }
}

// Row softmax over L_=512, scatter into out[:, t, :]
__global__ __launch_bounds__(128)
void k_softmax_out(const float* __restrict__ z2, float* __restrict__ out, int t) {
    __shared__ float red[128];
    const int b = blockIdx.x, tid = threadIdx.x;
    const float* row = z2 + b * L_;
    float v[4], mx = -3.402823466e38f;
#pragma unroll
    for (int i = 0; i < 4; ++i) { v[i] = row[tid + i * 128]; mx = fmaxf(mx, v[i]); }
    red[tid] = mx; __syncthreads();
    for (int s = 64; s > 0; s >>= 1) { if (tid < s) red[tid] = fmaxf(red[tid], red[tid + s]); __syncthreads(); }
    mx = red[0]; __syncthreads();
    float sum = 0.f;
#pragma unroll
    for (int i = 0; i < 4; ++i) { v[i] = __expf(v[i] - mx); sum += v[i]; }
    red[tid] = sum; __syncthreads();
    for (int s = 64; s > 0; s >>= 1) { if (tid < s) red[tid] += red[tid + s]; __syncthreads(); }
    float inv = 1.f / red[0];
    float* orow = out + ((long long)b * TD_ + t) * L_;
#pragma unroll
    for (int i = 0; i < 4; ++i) orow[tid + i * 128] = v[i] * inv;
}

// ---------------------------------------------------------------------------
extern "C" void kernel_launch(void* const* d_in, const int* in_sizes, int n_in,
                              void* d_out, int out_size, void* d_ws, size_t ws_size,
                              hipStream_t stream) {
    (void)in_sizes; (void)n_in; (void)out_size; (void)ws_size;
    const int*   tokens = (const int*)d_in[0];
    const float* emb    = (const float*)d_in[2];
    const float* Wk     = (const float*)d_in[3];
    const float* Wr     = (const float*)d_in[4];
    const float* bz     = (const float*)d_in[5];
    const float* Wm     = (const float*)d_in[6];
    const float* Wa     = (const float*)d_in[7];
    const float* Wo     = (const float*)d_in[8];
    const float* bo     = (const float*)d_in[9];
    const float* memory = (const float*)d_in[10];
    const float* h0     = (const float*)d_in[11];
    const float* c0     = (const float*)d_in[12];
    float* out = (float*)d_out;

    // Workspace carve (256B aligned)
    char* p = (char*)d_ws;
    auto carve = [&](size_t bytes) -> char* {
        char* q = p; p += (bytes + 255) & ~(size_t)255; return q;
    };
    bf16*  WzT   = (bf16*)carve((size_t)NZ * KZ * 2);          // [4096][2080]
    bf16*  WaT   = (bf16*)carve((size_t)D_ * KA * 2);          // [1024][2048]
    bf16*  WoT   = (bf16*)carve((size_t)L_ * D_ * 2);          // [512][1024]
    bf16*  WmT   = (bf16*)carve((size_t)D_ * ENC_ * 2);        // [1024][1024]
    bf16*  memb  = (bf16*)carve((size_t)B_ * TE_ * ENC_ * 2);  // bf16 memory
    float* keys  = (float*)carve((size_t)B_ * TE_ * D_ * 4);   // fp32 keys
    bf16*  xall  = (bf16*)carve((size_t)B_ * TD_ * E_ * 2);    // [t][b][32]
    bf16*  acat1 = (bf16*)carve((size_t)B_ * KZ * 2);          // [x|att|h] bf16
    bf16*  acat2 = (bf16*)carve((size_t)B_ * KA * 2);          // [h|ctx] bf16
    float* z     = (float*)carve((size_t)B_ * NZ * 4);
    float* hS    = (float*)carve((size_t)B_ * D_ * 4);
    float* cS    = (float*)carve((size_t)B_ * D_ * 4);
    float* z2    = (float*)carve((size_t)B_ * L_ * 4);

    auto cdiv = [](long long a, long long b) { return (int)((a + b - 1) / b); };

    // ---- one-time prep ----
    k_wz_tr<<<cdiv((long long)NZ * KZ, 256), 256, 0, stream>>>(Wk, Wr, WzT);
    k_tr_bf16<<<cdiv((long long)D_ * KA, 256), 256, 0, stream>>>(Wa, WaT, KA, D_, (long long)D_ * KA);
    k_tr_bf16<<<cdiv((long long)L_ * D_, 256), 256, 0, stream>>>(Wo, WoT, D_, L_, (long long)L_ * D_);
    k_tr_bf16<<<cdiv((long long)D_ * ENC_, 256), 256, 0, stream>>>(Wm, WmT, ENC_, D_, (long long)D_ * ENC_);
    k_cvt_bf16<<<cdiv((long long)B_ * TE_ * ENC_, 256), 256, 0, stream>>>(memory, memb, (long long)B_ * TE_ * ENC_);
    k_embed_all<<<cdiv(B_ * TD_ * E_, 256), 256, 0, stream>>>(tokens, emb, xall);
    k_init_state<<<cdiv(B_ * D_, 256), 256, 0, stream>>>(h0, c0, hS, cS, acat1);

    // keys[b] = memory[b] @ Wm : batched M=128 (8 M-tiles), N=1024, K=1024
    k_gemm_bf16<8><<<B_ * 8, 256, 0, stream>>>(
        memb, (long long)TE_ * ENC_, ENC_, nullptr, WmT,
        keys, (long long)TE_ * D_, D_, nullptr, 0, nullptr, ENC_, 8);

    // ---- recurrence ----
    for (int t = 0; t < TD_; ++t) {
        // z = [x_t|att|h] @ [Wk;Wr] + b : M=64, N=4096, K=2080
        // (x_t slice fused: kk==0 chunk sourced from xall[t])
        k_gemm_bf16<4><<<32, 256, 0, stream>>>(
            acat1, 0, KZ, xall + (long long)t * B_ * E_, WzT,
            z, 0, NZ, nullptr, 0, bz, KZ, 32);

        k_cell<<<cdiv(B_ * D_, 256), 256, 0, stream>>>(z, hS, cS, acat1);

        k_attention<<<B_, 128, 0, stream>>>(hS, keys, memory, acat2);

        // att = [h|ctx] @ Wa : M=64, N=1024, K=2048 -> bf16 into acat1 att-slice
        k_gemm_bf16<4><<<8, 256, 0, stream>>>(
            acat2, 0, KA, nullptr, WaT,
            nullptr, 0, 0, acat1 + E_, KZ, nullptr, KA, 8);

        // z2 = att @ Wo + bo : M=64, N=512, K=1024 (A = acat1 att-slice, lda=2080)
        k_gemm_bf16<4><<<4, 256, 0, stream>>>(
            acat1 + E_, 0, KZ, nullptr, WoT,
            z2, 0, L_, nullptr, 0, bo, D_, 4);

        k_softmax_out<<<B_, 128, 0, stream>>>(z2, out, t);
    }
}
